// AfmoeMoE_71442486002159
// MI455X (gfx1250) — compile-verified, matrix operand order checked by reading, not run on hardware
//
#include <hip/hip_runtime.h>
#include <hip/hip_bf16.h>

// ---------------- problem constants ----------------
#define TT   4096      // tokens
#define HH   1024      // hidden
#define EE   16        // experts
#define II   512       // routed intermediate
#define ISHH 1024      // shared intermediate
#define KTOP 2

typedef __attribute__((ext_vector_type(16))) __bf16 v16bf;
typedef __attribute__((ext_vector_type(8)))  float  v8f;

union Frag { uint4 q[2]; v16bf v; };

__device__ __forceinline__ unsigned short f2bf(float f) {
    unsigned int u = __builtin_bit_cast(unsigned int, f);
    u += 0x7FFFu + ((u >> 16) & 1u);   // round-to-nearest-even
    return (unsigned short)(u >> 16);
}
__device__ __forceinline__ unsigned int f2bf2(float lo, float hi) {
    return (unsigned int)f2bf(lo) | ((unsigned int)f2bf(hi) << 16);
}

// ---------------- router: sigmoid scores, top-2, normalized weights ----------------
__global__ __launch_bounds__(128) void router_kernel(
    const float* __restrict__ x, const float* __restrict__ rw,
    const float* __restrict__ bias, int* __restrict__ counts,
    int* __restrict__ list, float* __restrict__ wts)
{
    __shared__ float part[128];
    __shared__ float sc[EE];
    const int t   = blockIdx.x;
    const int tid = threadIdx.x;
    const int e   = tid >> 3, l = tid & 7;
    const float* xr = x + (size_t)t * HH;
    float s = 0.f;
    for (int h = l; h < HH; h += 8) s += xr[h] * rw[(size_t)h * EE + e];
    part[tid] = s;
    __syncthreads();
    if (tid < EE) {
        float v = 0.f;
        #pragma unroll
        for (int i = 0; i < 8; ++i) v += part[tid * 8 + i];
        sc[tid] = 1.f / (1.f + __expf(-v));
    }
    __syncthreads();
    if (tid == 0) {
        int b0 = -1, b1 = -1; float v0 = -1e30f, v1 = -1e30f;
        for (int i = 0; i < EE; ++i) {
            float v = sc[i] + bias[i];
            if (v > v0) { v1 = v0; b1 = b0; v0 = v; b0 = i; }
            else if (v > v1) { v1 = v; b1 = i; }
        }
        float w0 = sc[b0], w1 = sc[b1];
        float inv = 1.f / (w0 + w1 + 1e-20f);
        w0 *= inv; w1 *= inv;
        int s0 = atomicAdd(&counts[b0], 1);
        list[b0 * TT + s0] = t; wts[b0 * TT + s0] = w0;
        int s1 = atomicAdd(&counts[b1], 1);
        list[b1 * TT + s1] = t; wts[b1 * TT + s1] = w1;
    }
}

__global__ void zero_kernel(int* counts) { if (threadIdx.x < EE) counts[threadIdx.x] = 0; }

__global__ void scan_kernel(const int* __restrict__ counts, int* __restrict__ base) {
    if (threadIdx.x == 0) {
        int acc = 0;
        for (int e = 0; e < EE; ++e) { base[e] = acc; acc += counts[e]; }
    }
}

// ---------------- x fp32 -> bf16 (one-time; A operand reused by 4 GEMM passes) ----------------
__global__ __launch_bounds__(256) void cvt_kernel(const float* __restrict__ x,
                                                  unsigned short* __restrict__ xb) {
    int i = (blockIdx.x * 256 + threadIdx.x) * 4;   // grid sized for exact coverage
    float4 f = *(const float4*)(x + i);
    *(uint2*)(xb + i) = make_uint2(f2bf2(f.x, f.y), f2bf2(f.z, f.w));
}

// ---------------- fused gate+up GEMM (+SiLU) : act = silu(A@Wg) * (A@Wu) ----------------
// 64x64 block tile, 8 waves, 2 WMMA 16x16 tiles per wave per matrix, K-slab = 32,
// double-buffered LDS (one barrier per K-slab).
__global__ __launch_bounds__(256) void gemm_gateup(
    const unsigned short* __restrict__ A,       // bf16 [TT][HH]
    const float* __restrict__ WgB, const float* __restrict__ WuB,
    long wEStride, int N,
    const int* __restrict__ listB, const int* __restrict__ counts,
    const int* __restrict__ baseArr,
    unsigned short* __restrict__ actOut)        // bf16 [rows][N]
{
    const int e  = blockIdx.z;
    const int M  = counts ? counts[e] : TT;
    const int m0 = blockIdx.y * 64;
    if (m0 >= M) return;
    const int n0      = blockIdx.x * 64;
    const int outBase = baseArr ? baseArr[e] : 0;
    const int* rows   = listB ? listB + e * TT : nullptr;
    const float* Wg   = WgB + (size_t)e * wEStride;
    const float* Wu   = WuB + (size_t)e * wEStride;

    __shared__ unsigned short As [2][64][48];
    __shared__ unsigned short Bgs[2][64][48];   // transposed: [n][k]
    __shared__ unsigned short Bus[2][64][48];
    __shared__ int rowTok[64];

    const int tid = threadIdx.x;
    if (tid < 64) {
        int m = m0 + tid;
        rowTok[tid] = (m < M) ? (rows ? rows[m] : m) : -1;
    }
    __syncthreads();

    const int lane = tid & 31;
    const int wid  = tid >> 5;
    const int sm   = wid & 3;            // M subtile 0..3
    const int sn   = (wid >> 2) * 2;     // two N subtiles
    const int lr   = lane & 15;
    const int kh   = (lane >> 4) * 8;    // K-half select (bf16 A/B lane layout)

    v8f cg0 = {}; v8f cg1 = {}; v8f cu0 = {}; v8f cu1 = {};

    const int ar  = tid >> 2;            // A: row 0..63
    const int ap  = (tid & 3) * 8;       // A: 8 contiguous bf16
    const int bn  = tid & 63;            // B: col (fast index -> coalesced)
    const int bkq = (tid >> 6) * 8;      // B: 8 consecutive k per thread

    auto stageA = [&](int k0, int buf) {
        int tok = rowTok[ar];
        uint4 v = {0, 0, 0, 0};
        if (tok >= 0) v = *(const uint4*)(A + (size_t)tok * HH + k0 + ap);
        *(uint4*)&As[buf][ar][ap] = v;
    };
    auto stageB = [&](int k0, int buf) {
        const float* g = Wg + (size_t)(k0 + bkq) * N + n0 + bn;
        const float* u = Wu + (size_t)(k0 + bkq) * N + n0 + bn;
        if (k0 + 32 < HH) {   // prefetch the following K-slab of the streamed operands
            __builtin_prefetch(g + (size_t)32 * N, 0, 0);
            __builtin_prefetch(u + (size_t)32 * N, 0, 0);
        }
        float gv[8], uv[8];
        #pragma unroll
        for (int j = 0; j < 8; ++j) { gv[j] = g[(size_t)j * N]; uv[j] = u[(size_t)j * N]; }
        uint4 qg = { f2bf2(gv[0], gv[1]), f2bf2(gv[2], gv[3]), f2bf2(gv[4], gv[5]), f2bf2(gv[6], gv[7]) };
        uint4 qu = { f2bf2(uv[0], uv[1]), f2bf2(uv[2], uv[3]), f2bf2(uv[4], uv[5]), f2bf2(uv[6], uv[7]) };
        *(uint4*)&Bgs[buf][bn][bkq] = qg;   // one ds_store_b128 per matrix
        *(uint4*)&Bus[buf][bn][bkq] = qu;
    };

    stageA(0, 0); stageB(0, 0);
    int buf = 0;
    for (int k0 = 0; k0 < HH; k0 += 32) {
        __syncthreads();                         // buffer `buf` ready
        if (k0 + 32 < HH) { stageA(k0 + 32, buf ^ 1); stageB(k0 + 32, buf ^ 1); }

        Frag fa, fg0, fg1, fu0, fu1;
        fa.q[0]  = *(const uint4*)&As [buf][sm * 16 + lr][kh];
        fa.q[1]  = *(const uint4*)&As [buf][sm * 16 + lr][16 + kh];
        fg0.q[0] = *(const uint4*)&Bgs[buf][(sn + 0) * 16 + lr][kh];
        fg0.q[1] = *(const uint4*)&Bgs[buf][(sn + 0) * 16 + lr][16 + kh];
        fg1.q[0] = *(const uint4*)&Bgs[buf][(sn + 1) * 16 + lr][kh];
        fg1.q[1] = *(const uint4*)&Bgs[buf][(sn + 1) * 16 + lr][16 + kh];
        fu0.q[0] = *(const uint4*)&Bus[buf][(sn + 0) * 16 + lr][kh];
        fu0.q[1] = *(const uint4*)&Bus[buf][(sn + 0) * 16 + lr][16 + kh];
        fu1.q[0] = *(const uint4*)&Bus[buf][(sn + 1) * 16 + lr][kh];
        fu1.q[1] = *(const uint4*)&Bus[buf][(sn + 1) * 16 + lr][16 + kh];

        cg0 = __builtin_amdgcn_wmma_f32_16x16x32_bf16(false, fa.v, false, fg0.v, (short)0, cg0, false, false);
        cg1 = __builtin_amdgcn_wmma_f32_16x16x32_bf16(false, fa.v, false, fg1.v, (short)0, cg1, false, false);
        cu0 = __builtin_amdgcn_wmma_f32_16x16x32_bf16(false, fa.v, false, fu0.v, (short)0, cu0, false, false);
        cu1 = __builtin_amdgcn_wmma_f32_16x16x32_bf16(false, fa.v, false, fu1.v, (short)0, cu1, false, false);
        buf ^= 1;
    }

    // ---- epilogue: SiLU(g)*u -> bf16 ----
    #pragma unroll
    for (int i = 0; i < 2; ++i) {
        v8f g = i ? cg1 : cg0;
        v8f u = i ? cu1 : cu0;
        int col = n0 + (sn + i) * 16 + lr;
        #pragma unroll
        for (int r = 0; r < 8; ++r) {
            int m = m0 + sm * 16 + r + kh;     // C/D layout: VGPR r -> M=r (+8 for lanes 16..31)
            if (m < M) {
                float gv = g[r], uv = u[r];
                float a = gv / (1.f + __expf(-gv)) * uv;
                actOut[(size_t)(outBase + m) * N + col] = f2bf(a);
            }
        }
    }
}

// ---------------- down GEMM : out[tok] (+)= wt * (act @ Wd) ----------------
__global__ __launch_bounds__(256) void gemm_down(
    const unsigned short* __restrict__ Aact,    // bf16 [rows][Nin]
    const float* __restrict__ WdB, long wEStride, int Nin,
    const int* __restrict__ listB, const float* __restrict__ wtB,
    const int* __restrict__ counts, const int* __restrict__ baseArr,
    float* __restrict__ out, int isRouted)
{
    const int e  = blockIdx.z;
    const int M  = counts ? counts[e] : TT;
    const int m0 = blockIdx.y * 64;
    if (m0 >= M) return;
    const int n0    = blockIdx.x * 64;          // over H
    const int aBase = baseArr ? baseArr[e] : 0;
    const float* Wd = WdB + (size_t)e * wEStride;
    const int*   rows = listB ? listB + e * TT : nullptr;
    const float* wts  = wtB  ? wtB  + e * TT : nullptr;

    __shared__ unsigned short As[2][64][48];
    __shared__ unsigned short Bs[2][64][48];

    const int tid  = threadIdx.x;
    const int lane = tid & 31;
    const int wid  = tid >> 5;
    const int sm   = wid & 3;
    const int sn   = (wid >> 2) * 2;
    const int lr   = lane & 15;
    const int kh   = (lane >> 4) * 8;

    v8f c0 = {}; v8f c1 = {};

    const int ar  = tid >> 2;
    const int ap  = (tid & 3) * 8;
    const int bn  = tid & 63;
    const int bkq = (tid >> 6) * 8;

    auto stageA = [&](int k0, int buf) {
        int m = m0 + ar;
        uint4 v = {0, 0, 0, 0};
        if (m < M) v = *(const uint4*)(Aact + (size_t)(aBase + m) * Nin + k0 + ap);
        *(uint4*)&As[buf][ar][ap] = v;
    };
    auto stageB = [&](int k0, int buf) {
        const float* w = Wd + (size_t)(k0 + bkq) * HH + n0 + bn;
        if (k0 + 32 < Nin) __builtin_prefetch(w + (size_t)32 * HH, 0, 0);
        float wv[8];
        #pragma unroll
        for (int j = 0; j < 8; ++j) wv[j] = w[(size_t)j * HH];
        uint4 q = { f2bf2(wv[0], wv[1]), f2bf2(wv[2], wv[3]), f2bf2(wv[4], wv[5]), f2bf2(wv[6], wv[7]) };
        *(uint4*)&Bs[buf][bn][bkq] = q;
    };

    stageA(0, 0); stageB(0, 0);
    int buf = 0;
    for (int k0 = 0; k0 < Nin; k0 += 32) {
        __syncthreads();
        if (k0 + 32 < Nin) { stageA(k0 + 32, buf ^ 1); stageB(k0 + 32, buf ^ 1); }

        Frag fa, fb0, fb1;
        fa.q[0]  = *(const uint4*)&As[buf][sm * 16 + lr][kh];
        fa.q[1]  = *(const uint4*)&As[buf][sm * 16 + lr][16 + kh];
        fb0.q[0] = *(const uint4*)&Bs[buf][(sn + 0) * 16 + lr][kh];
        fb0.q[1] = *(const uint4*)&Bs[buf][(sn + 0) * 16 + lr][16 + kh];
        fb1.q[0] = *(const uint4*)&Bs[buf][(sn + 1) * 16 + lr][kh];
        fb1.q[1] = *(const uint4*)&Bs[buf][(sn + 1) * 16 + lr][16 + kh];

        c0 = __builtin_amdgcn_wmma_f32_16x16x32_bf16(false, fa.v, false, fb0.v, (short)0, c0, false, false);
        c1 = __builtin_amdgcn_wmma_f32_16x16x32_bf16(false, fa.v, false, fb1.v, (short)0, c1, false, false);
        buf ^= 1;
    }

    #pragma unroll
    for (int i = 0; i < 2; ++i) {
        v8f c = i ? c1 : c0;
        int col = n0 + (sn + i) * 16 + lr;
        #pragma unroll
        for (int r = 0; r < 8; ++r) {
            int m = m0 + sm * 16 + r + kh;
            if (m < M) {
                if (isRouted) {
                    int tok = rows[m];
                    atomicAdd(&out[(size_t)tok * HH + col], wts[m] * c[r]);
                } else {
                    out[(size_t)m * HH + col] = c[r];   // shared pass: full plain-store coverage
                }
            }
        }
    }
}

// ---------------- host launch ----------------
extern "C" void kernel_launch(void* const* d_in, const int* in_sizes, int n_in,
                              void* d_out, int out_size, void* d_ws, size_t ws_size,
                              hipStream_t stream) {
    const float* x    = (const float*)d_in[0];
    const float* rw   = (const float*)d_in[1];
    const float* bias = (const float*)d_in[2];
    const float* wg   = (const float*)d_in[3];
    const float* wu   = (const float*)d_in[4];
    const float* wd   = (const float*)d_in[5];
    const float* swg  = (const float*)d_in[6];
    const float* swu  = (const float*)d_in[7];
    const float* swd  = (const float*)d_in[8];
    float* out = (float*)d_out;

    char* ws = (char*)d_ws;
    // workspace layout (~26 MB)
    int*            counts = (int*)(ws + 0);
    int*            base   = (int*)(ws + 256);
    int*            list   = (int*)(ws + 512);                                   // EE*TT ints
    float*          wts    = (float*)(ws + 512 + (size_t)EE * TT * 4);           // EE*TT floats
    unsigned short* xb     = (unsigned short*)(ws + 512 + (size_t)EE * TT * 8);  // TT*HH bf16
    unsigned short* act    = xb  + (size_t)TT * HH;                              // TT*KTOP*II bf16
    unsigned short* sact   = act + (size_t)TT * KTOP * II;                       // TT*ISHH bf16

    zero_kernel  <<<1, 32, 0, stream>>>(counts);
    router_kernel<<<TT, 128, 0, stream>>>(x, rw, bias, counts, list, wts);
    scan_kernel  <<<1, 32, 0, stream>>>(counts, base);
    cvt_kernel   <<<(TT * HH) / (256 * 4), 256, 0, stream>>>(x, xb);

    // shared expert (dense rows; writes out with plain stores)
    gemm_gateup<<<dim3(ISHH / 64, TT / 64, 1), 256, 0, stream>>>(
        xb, swg, swu, 0L, ISHH, nullptr, nullptr, nullptr, sact);
    gemm_down  <<<dim3(HH / 64, TT / 64, 1), 256, 0, stream>>>(
        sact, swd, 0L, ISHH, nullptr, nullptr, nullptr, nullptr, out, 0);

    // routed experts (gathered rows; scatter-accumulate after shared wrote out)
    gemm_gateup<<<dim3(II / 64, TT / 64, EE), 256, 0, stream>>>(
        xb, wg, wu, (long)HH * II, II, list, counts, base, act);
    gemm_down  <<<dim3(HH / 64, TT / 64, EE), 256, 0, stream>>>(
        act, wd, (long)II * HH, II, list, wts, counts, base, out, 1);
}
